// DFFN_9904194585031
// MI455X (gfx1250) — compile-verified, matrix operand order checked by reading, not run on hardware
//
#include <hip/hip_runtime.h>
#include <hip/hip_bf16.h>

typedef __attribute__((ext_vector_type(16))) _Float16 v16h;
typedef __attribute__((ext_vector_type(2)))  _Float16 h2;
typedef __attribute__((ext_vector_type(8)))  float    v8f;
typedef __attribute__((ext_vector_type(4)))  float    f4;

#define B_     4
#define C_     64
#define H_     256
#define W_     256
#define HID_   170
#define TILE_  16
#define EXT_   20          // TILE + 2*2 halo (patch-aligned)
#define NPIX_  (EXT_*EXT_) // 400
#define XPAD_  72          // padded channel stride (16B-aligned, bank-friendly)
#define NGRP_  6           // ceil(170/32) groups of 32 gathered hidden channels
#define WROWS_ 192         // 6*32 gathered w_in rows (zero-padded)
#define HFPAD_ 404         // padded pixel stride for filtered-h tile
#define WOPAD_ 200         // padded hidden stride for w_out (cols >=170 zero)

#if defined(__HIP_DEVICE_COMPILE__) && defined(__gfx1250__)
#define USE_ASYNC_LDS 1
#else
#define USE_ASYNC_LDS 0
#endif

#if USE_ASYNC_LDS
// byte offset of a generic pointer within the wave's LDS aperture
__device__ static inline unsigned lds_offset_of(const void* p) {
  return (unsigned)(unsigned long long)
         (__attribute__((address_space(3))) const char*)p;
}
#endif

__global__ __launch_bounds__(256) void dffn_fused_kernel(
    const float* __restrict__ xin,  const float* __restrict__ w_in,
    const float* __restrict__ fft_w, const float* __restrict__ w_dw,
    const float* __restrict__ w_out, float* __restrict__ out)
{
  __shared__ __align__(16) _Float16 x_lds  [NPIX_ * XPAD_];   // 57.6 KB
  __shared__ __align__(16) _Float16 win_lds[WROWS_ * XPAD_];  // 27.6 KB
  __shared__ __align__(16) _Float16 hf_lds [32 * HFPAD_];     // 25.9 KB
  __shared__ __align__(16) _Float16 wout_lds[64 * WOPAD_];    // 25.6 KB
  __shared__ __align__(16) float    wdw_lds [2*HID_*9];       // 12.2 KB
  __shared__ __align__(16) float    fftw_lds[HID_*4];         //  2.7 KB

  const int tid  = threadIdx.x;
  const int wave = tid >> 5;
  const int nm   = tid & 15;        // M (pixel) or N (channel) index within 16
  const int hh   = (tid >> 4) & 1;  // wave half -> K-range selector in A/B layouts

  const int tx0 = blockIdx.x * TILE_;
  const int ty0 = blockIdx.y * TILE_;
  const int b   = blockIdx.z;

  const float* xb = xin + (size_t)b * C_ * H_ * W_;

  // ---- kick off async DMA of the pure-copy weight tables into LDS ----
  // (tracked by ASYNCcnt; overlapped with x-tile staging below)
#if USE_ASYNC_LDS
  for (int i = tid; i < (2*HID_*9)/4; i += 256) {       // 765 x b128
    unsigned ldst = lds_offset_of(&wdw_lds[i*4]);
    unsigned long long ga = (unsigned long long)(const void*)(w_dw + i*4);
    asm volatile("global_load_async_to_lds_b128 %0, %1, off"
                 :: "v"(ldst), "v"(ga) : "memory");
  }
  for (int i = tid; i < (HID_*4)/4; i += 256) {         // 170 x b128
    unsigned ldst = lds_offset_of(&fftw_lds[i*4]);
    unsigned long long ga = (unsigned long long)(const void*)(fft_w + i*4);
    asm volatile("global_load_async_to_lds_b128 %0, %1, off"
                 :: "v"(ldst), "v"(ga) : "memory");
  }
#else
  for (int i = tid; i < 2*HID_*9; i += 256) wdw_lds[i]  = w_dw[i];
  for (int i = tid; i < HID_*4;   i += 256) fftw_lds[i] = fft_w[i];
#endif

  // ---- prefetch the x rows this block is about to stage (global_prefetch_b8)
  for (int idx = tid; idx < C_ * EXT_; idx += 256) {
    int c = idx / EXT_, py = idx - c * EXT_;
    int gy = ty0 - 2 + py;
    int gx = tx0 - 2; gx = gx < 0 ? 0 : gx;
    if ((unsigned)gy < (unsigned)H_)
      __builtin_prefetch(&xb[(size_t)c * (H_*W_) + gy * W_ + gx], 0, 3);
  }

  // ---------------- stage converted operands into LDS ----------------
  for (int idx = tid; idx < C_ * NPIX_; idx += 256) {
    int c = idx / NPIX_;
    int p = idx - c * NPIX_;
    int py = p / EXT_, px = p - py * EXT_;
    int gy = ty0 - 2 + py, gx = tx0 - 2 + px;
    float v = 0.0f;
    if ((unsigned)gy < (unsigned)H_ && (unsigned)gx < (unsigned)W_)
      v = xb[(size_t)c * (H_*W_) + gy * W_ + gx];
    x_lds[p * XPAD_ + c] = (_Float16)v;   // zero halo => exact zero FFT patches
  }
  // gathered project_in rows: group j, slot s<16 -> ch 16j+s (low half),
  // slot s>=16 -> ch 85+16j+(s-16) (high half); invalid slots zero-padded
  for (int idx = tid; idx < WROWS_ * C_; idx += 256) {
    int row = idx >> 6;
    int c   = idx & 63;
    int j = row >> 5, s = row & 31, t = s & 15;
    int hch = ((s < 16) ? 0 : 85) + 16*j + t;
    float v = (16*j + t < 85) ? w_in[hch * C_ + c] : 0.0f;
    win_lds[row * XPAD_ + c] = (_Float16)v;
  }
  for (int idx = tid; idx < 64 * WOPAD_; idx += 256) {
    int r = idx / WOPAD_, o = idx - r * WOPAD_;
    float v = (o < HID_) ? w_out[r * HID_ + o] : 0.0f;
    wout_lds[idx] = (_Float16)v;
  }

#if USE_ASYNC_LDS
  asm volatile("s_wait_asynccnt 0" ::: "memory");
#endif
  __syncthreads();

  // persistent project_out accumulators: 2 pixel-blocks x 4 out-channel tiles
  v8f acc2[2][4];
  #pragma unroll
  for (int q = 0; q < 2; ++q)
    #pragma unroll
    for (int n = 0; n < 4; ++n) acc2[q][n] = (v8f){0,0,0,0,0,0,0,0};

  for (int j = 0; j < NGRP_; ++j) {
    // ---- GEMM1 (project_in) over extended tile + 2x2 Hadamard FFT filter ----
    // 25 pixel-blocks (5x5 of 4x4) x 2 channel-ntiles = 50 wave-jobs
    for (int id = wave; id < 50; id += 8) {
      int nt = id & 1;
      int mb = id >> 1;
      int mby = mb / 5, mbx = mb - mby * 5;
      int py = mby*4 + (nm >> 2);
      int px = mbx*4 + (nm & 3);
      const _Float16* ap = &x_lds[(py*EXT_ + px) * XPAD_ + 8*hh];
      const _Float16* bp = &win_lds[(j*32 + nt*16 + nm) * XPAD_ + 8*hh];
      v16h A0, A1, B0, B1;
      ((f4*)&A0)[0] = *(const f4*)(ap);        // K  8h.. 8h+7
      ((f4*)&A0)[1] = *(const f4*)(ap + 16);   // K 16+8h..
      ((f4*)&A1)[0] = *(const f4*)(ap + 32);
      ((f4*)&A1)[1] = *(const f4*)(ap + 48);
      ((f4*)&B0)[0] = *(const f4*)(bp);
      ((f4*)&B0)[1] = *(const f4*)(bp + 16);
      ((f4*)&B1)[0] = *(const f4*)(bp + 32);
      ((f4*)&B1)[1] = *(const f4*)(bp + 48);
      v8f acc = (v8f){0,0,0,0,0,0,0,0};
      acc = __builtin_amdgcn_wmma_f32_16x16x32_f16(false, A0, false, B0, (short)0, acc, false, false);
      acc = __builtin_amdgcn_wmma_f32_16x16x32_f16(false, A1, false, B1, (short)0, acc, false, false);

      // per-lane channel's 2x2 spectral weights
      int sg  = nt*16 + nm;
      int hch = ((sg < 16) ? 0 : 85) + 16*j + nm;
      hch = (hch > HID_-1) ? (HID_-1) : hch;     // pad slots: values are 0 anyway
      float w00 = fftw_lds[hch*4+0], w01 = fftw_lds[hch*4+1];
      float w10 = fftw_lds[hch*4+2], w11 = fftw_lds[hch*4+3];
      // each lane holds both rows of its 2 patches (C-layout rows 2h,2h+1)
      #pragma unroll
      for (int pc = 0; pc < 2; ++pc) {
        float s00 = acc[2*pc],   s01 = acc[2*pc+1];
        float s10 = acc[4+2*pc], s11 = acc[4+2*pc+1];
        float F00 = (s00+s01+s10+s11)*w00;
        float F01 = (s00-s01+s10-s11)*w01;
        float F10 = (s00+s01-s10-s11)*w10;
        float F11 = (s00-s01-s10+s11)*w11;
        float y00 = 0.25f*(F00+F01+F10+F11);
        float y01 = 0.25f*(F00-F01+F10-F11);
        float y10 = 0.25f*(F00+F01-F10-F11);
        float y11 = 0.25f*(F00-F01-F10+F11);
        int base0 = sg*HFPAD_ + (mby*4 + 2*hh)*EXT_ + (mbx*4 + 2*pc);
        h2 v0; v0[0] = (_Float16)y00; v0[1] = (_Float16)y01;
        h2 v1; v1[0] = (_Float16)y10; v1[1] = (_Float16)y11;
        *(h2*)&hf_lds[base0]        = v0;
        *(h2*)&hf_lds[base0 + EXT_] = v1;
      }
    }
    __syncthreads();

    // ---- depthwise 3x3 + exact-GELU gate, built directly in GEMM2 A-layout ----
    #pragma unroll
    for (int q = 0; q < 2; ++q) {
      int mb  = wave*2 + q;                 // 16 blocks over 16x16 tile
      int mby = mb >> 2, mbx = mb & 3;
      int oy = mby*4 + (nm >> 2), ox = mbx*4 + (nm & 3);
      int pcen = (oy + 2)*EXT_ + (ox + 2);  // center in extended coords
      _Float16 a2h[16];
      #pragma unroll
      for (int tt = 0; tt < 8; ++tt) {
        int t = (tt < 4) ? (4*hh + tt) : (4 + 4*hh + tt);   // lane's hf slot (low half)
        const _Float16* pl = &hf_lds[t*HFPAD_      + pcen - EXT_ - 1];
        const _Float16* ph = &hf_lds[(16+t)*HFPAD_ + pcen - EXT_ - 1];
        float lo[9], hi[9];
        #pragma unroll
        for (int r = 0; r < 3; ++r)
          #pragma unroll
          for (int cx = 0; cx < 3; ++cx) {
            lo[r*3+cx] = (float)pl[r*EXT_ + cx];
            hi[r*3+cx] = (float)ph[r*EXT_ + cx];
          }
        #pragma unroll
        for (int u = 0; u < 2; ++u) {
          int k = 2*t + u;
          int c = 32*j + k;                 // gated hidden channel
          float g = 0.0f;
          if (c < HID_) {
            const float* wl = &wdw_lds[c*9];
            const float* wh = &wdw_lds[(c+HID_)*9];
            float s1 = 0.0f, s2 = 0.0f;
            #pragma unroll
            for (int i = 0; i < 9; ++i) {
              s1 = fmaf(lo[i], wl[i], s1);
              s2 = fmaf(hi[i], wh[i], s2);
            }
            float ge = 0.5f * s1 * (1.0f + erff(s1 * 0.70710678118654752f));
            g = ge * s2;
          }
          a2h[2*tt + u] = (_Float16)g;      // matches A-layout K index for this lane
        }
      }
      v16h A2;
      #pragma unroll
      for (int e = 0; e < 16; ++e) A2[e] = a2h[e];
      #pragma unroll
      for (int n2 = 0; n2 < 4; ++n2) {
        const _Float16* bp = &wout_lds[(n2*16 + nm) * WOPAD_ + 32*j + 8*hh];
        v16h B2;
        ((f4*)&B2)[0] = *(const f4*)(bp);
        ((f4*)&B2)[1] = *(const f4*)(bp + 16);
        acc2[q][n2] = __builtin_amdgcn_wmma_f32_16x16x32_f16(
            false, A2, false, B2, (short)0, acc2[q][n2], false, false);
      }
    }
    __syncthreads();
  }

  // ---------------- epilogue: f32 output, float4 rows ----------------
  float* ob = out + (size_t)b * C_ * H_ * W_;
  #pragma unroll
  for (int q = 0; q < 2; ++q) {
    int mb  = wave*2 + q;
    int mby = mb >> 2, mbx = mb & 3;
    int gy0 = ty0 + mby*4 + 2*hh;
    int gx0 = tx0 + mbx*4;
    #pragma unroll
    for (int n2 = 0; n2 < 4; ++n2) {
      int cch = n2*16 + nm;
      v8f a = acc2[q][n2];
      f4 r0 = (f4){a[0], a[1], a[2], a[3]};
      f4 r1 = (f4){a[4], a[5], a[6], a[7]};
      size_t off = (size_t)cch * (H_*W_) + (size_t)gy0 * W_ + gx0;
      *(f4*)&ob[off]      = r0;
      *(f4*)&ob[off + W_] = r1;
    }
  }
}

extern "C" void kernel_launch(void* const* d_in, const int* in_sizes, int n_in,
                              void* d_out, int out_size, void* d_ws, size_t ws_size,
                              hipStream_t stream) {
  (void)in_sizes; (void)n_in; (void)d_ws; (void)ws_size; (void)out_size;
  const float* x     = (const float*)d_in[0];
  const float* w_in  = (const float*)d_in[1];
  const float* fft_w = (const float*)d_in[2];
  const float* w_dw  = (const float*)d_in[3];
  const float* w_out = (const float*)d_in[4];
  float* out = (float*)d_out;
  dim3 grid(W_/TILE_, H_/TILE_, B_);
  dffn_fused_kernel<<<grid, 256, 0, stream>>>(x, w_in, fft_w, w_dw, w_out, out);
}
